// BayesianLinear_28011776705085
// MI455X (gfx1250) — compile-verified, matrix operand order checked by reading, not run on hardware
//
#include <hip/hip_runtime.h>
#include <math.h>

#define B_SZ   4096
#define IN_SZ  2048
#define OUT_SZ 2048

#define BM 128
#define BN 64
#define BK 32
#define LDA 40            // padded LDS row stride (halfs): 80B, 16B-aligned, bank-spread
#define NKIT (IN_SZ / BK)

typedef __attribute__((ext_vector_type(16))) __bf16 v16bf;
typedef __attribute__((ext_vector_type(8)))  __bf16 v8bf;
typedef __attribute__((ext_vector_type(8)))  float  v8f;
typedef __attribute__((ext_vector_type(4)))  float  v4f;

union Frag16 { v16bf v; v8bf h[2]; };

// ---------------- Pass 1a: X -> bf16 X, bf16 X^2 ----------------
__global__ void bayes_prep_x(const float* __restrict__ x,
                             __bf16* __restrict__ xb,
                             __bf16* __restrict__ x2b) {
  const size_t i = ((size_t)blockIdx.x * blockDim.x + threadIdx.x) * 8;
  v4f a = *(const v4f*)(x + i);
  v4f b = *(const v4f*)(x + i + 4);
  v8bf xo, x2o;
#pragma unroll
  for (int j = 0; j < 4; ++j) {
    xo[j]      = (__bf16)a[j];
    xo[j + 4]  = (__bf16)b[j];
    x2o[j]     = (__bf16)(a[j] * a[j]);
    x2o[j + 4] = (__bf16)(b[j] * b[j]);
  }
  *(v8bf*)(xb + i)  = xo;
  *(v8bf*)(x2b + i) = x2o;
}

// ------- Pass 1b: reparam sample W = mu + e^ls * eps, S = e^{2 ls} -------
__global__ void bayes_prep_w(const float* __restrict__ mu,
                             const float* __restrict__ ls,
                             const float* __restrict__ eps,
                             __bf16* __restrict__ wb,
                             __bf16* __restrict__ sb) {
  const size_t i = ((size_t)blockIdx.x * blockDim.x + threadIdx.x) * 8;
  v8bf wo, so;
#pragma unroll
  for (int c = 0; c < 2; ++c) {
    v4f m = *(const v4f*)(mu + i + 4 * c);
    v4f l = *(const v4f*)(ls + i + 4 * c);
    v4f e = *(const v4f*)(eps + i + 4 * c);
#pragma unroll
    for (int j = 0; j < 4; ++j) {
      const float sg = __expf(l[j]);
      wo[4 * c + j] = (__bf16)(m[j] + sg * e[j]);
      so[4 * c + j] = (__bf16)(sg * sg);
    }
  }
  *(v8bf*)(wb + i) = wo;
  *(v8bf*)(sb + i) = so;
}

// ---------------- Pass 2: fused dual GEMM on bf16 WMMA ----------------
__global__ __launch_bounds__(256)
void bayes_gemm(const __bf16* __restrict__ Xb,
                const __bf16* __restrict__ X2b,
                const __bf16* __restrict__ Wb,
                const __bf16* __restrict__ Sb,
                const float* __restrict__ bias_mu,
                const float* __restrict__ bias_ls,
                const float* __restrict__ eps_b,
                float* __restrict__ out,
                float* __restrict__ unc) {
  __shared__ __align__(16) __bf16 sX [2][BM * LDA];
  __shared__ __align__(16) __bf16 sX2[2][BM * LDA];
  __shared__ __align__(16) __bf16 sW [2][BN * LDA];
  __shared__ __align__(16) __bf16 sS [2][BN * LDA];

  const int tid  = threadIdx.x;
  const int lane = tid & 31;
  const int sel  = lane >> 4;     // 0: K chunk base 0, 1: base 8
  const int lr   = lane & 15;     // row/col within 16x16 tile
  const int wid  = tid >> 5;      // 8 waves
  const int wm   = wid >> 1;      // 0..3 : 32-row slice of block
  const int wn   = wid & 1;       // 0..1 : 32-col slice of block
  const int bm0  = blockIdx.y * BM;
  const int bn0  = blockIdx.x * BN;

  // global staging coords: X tiles 128x32 (2 thr/row), W tiles 64x32 (4 thr/row)
  const int xr = tid >> 1;
  const int xc = (tid & 1) * 16;
  const int wr = tid >> 2;
  const int wc = (tid & 3) * 8;

  const __bf16* gX  = Xb  + (size_t)(bm0 + xr) * IN_SZ + xc;
  const __bf16* gX2 = X2b + (size_t)(bm0 + xr) * IN_SZ + xc;
  const __bf16* gW  = Wb  + (size_t)(bn0 + wr) * IN_SZ + wc;
  const __bf16* gS  = Sb  + (size_t)(bn0 + wr) * IN_SZ + wc;

  v8f accO[2][2] = {{{0.f}}};
  v8f accU[2][2] = {{{0.f}}};

  // prologue: fill stage 0
  {
    v8bf x0 = *(const v8bf*)(gX);
    v8bf x1 = *(const v8bf*)(gX + 8);
    v8bf q0 = *(const v8bf*)(gX2);
    v8bf q1 = *(const v8bf*)(gX2 + 8);
    v8bf w0 = *(const v8bf*)(gW);
    v8bf s0 = *(const v8bf*)(gS);
    *(v8bf*)(&sX [0][xr * LDA + xc])     = x0;
    *(v8bf*)(&sX [0][xr * LDA + xc + 8]) = x1;
    *(v8bf*)(&sX2[0][xr * LDA + xc])     = q0;
    *(v8bf*)(&sX2[0][xr * LDA + xc + 8]) = q1;
    *(v8bf*)(&sW [0][wr * LDA + wc])     = w0;
    *(v8bf*)(&sS [0][wr * LDA + wc])     = s0;
  }
  __syncthreads();

  int cur = 0;
  for (int kt = 0; kt < NKIT; ++kt) {
    // issue next-stage global loads first (overlap with WMMA pipe)
    v8bf px0, px1, p20, p21, pw, ps;
    const bool pf = (kt + 1) < NKIT;
    if (pf) {
      const int ko = (kt + 1) * BK;
      px0 = *(const v8bf*)(gX + ko);
      px1 = *(const v8bf*)(gX + ko + 8);
      p20 = *(const v8bf*)(gX2 + ko);
      p21 = *(const v8bf*)(gX2 + ko + 8);
      pw  = *(const v8bf*)(gW + ko);
      ps  = *(const v8bf*)(gS + ko);
    }

    // load fragments from LDS (two 16B ds_load_b128 per fragment)
    Frag16 a[2], a2[2], bw[2], bs[2];
#pragma unroll
    for (int t = 0; t < 2; ++t) {
      const int ar = wm * 32 + t * 16 + lr;
      const int o0 = ar * LDA + sel * 8;
      a [t].h[0] = *(const v8bf*)(&sX [cur][o0]);
      a [t].h[1] = *(const v8bf*)(&sX [cur][o0 + 16]);
      a2[t].h[0] = *(const v8bf*)(&sX2[cur][o0]);
      a2[t].h[1] = *(const v8bf*)(&sX2[cur][o0 + 16]);
      const int br = wn * 32 + t * 16 + lr;
      const int p0 = br * LDA + sel * 8;
      bw[t].h[0] = *(const v8bf*)(&sW[cur][p0]);
      bw[t].h[1] = *(const v8bf*)(&sW[cur][p0 + 16]);
      bs[t].h[0] = *(const v8bf*)(&sS[cur][p0]);
      bs[t].h[1] = *(const v8bf*)(&sS[cur][p0 + 16]);
    }

#pragma unroll
    for (int mt = 0; mt < 2; ++mt)
#pragma unroll
      for (int nt = 0; nt < 2; ++nt) {
        accO[mt][nt] = __builtin_amdgcn_wmma_f32_16x16x32_bf16(
            false, a[mt].v, false, bw[nt].v, (short)0, accO[mt][nt], false, false);
        accU[mt][nt] = __builtin_amdgcn_wmma_f32_16x16x32_bf16(
            false, a2[mt].v, false, bs[nt].v, (short)0, accU[mt][nt], false, false);
      }

    if (pf) {
      const int nxt = cur ^ 1;
      *(v8bf*)(&sX [nxt][xr * LDA + xc])     = px0;
      *(v8bf*)(&sX [nxt][xr * LDA + xc + 8]) = px1;
      *(v8bf*)(&sX2[nxt][xr * LDA + xc])     = p20;
      *(v8bf*)(&sX2[nxt][xr * LDA + xc + 8]) = p21;
      *(v8bf*)(&sW [nxt][wr * LDA + wc])     = pw;
      *(v8bf*)(&sS [nxt][wr * LDA + wc])     = ps;
    }
    __syncthreads();
    cur ^= 1;
  }

  // epilogue: sampled bias + sqrt(var), write both outputs
#pragma unroll
  for (int nt = 0; nt < 2; ++nt) {
    const int col  = bn0 + wn * 32 + nt * 16 + lr;
    const float bl = bias_ls[col];
    const float bg = __expf(bl);
    const float badd = bias_mu[col] + bg * eps_b[col];
    const float bvar = bg * bg;
#pragma unroll
    for (int mt = 0; mt < 2; ++mt) {
      const int row0 = bm0 + wm * 32 + mt * 16 + sel * 8;
#pragma unroll
      for (int r = 0; r < 8; ++r) {
        const size_t idx = (size_t)(row0 + r) * OUT_SZ + col;
        out[idx] = accO[mt][nt][r] + badd;
        unc[idx] = sqrtf(accU[mt][nt][r] + bvar);
      }
    }
  }
}

extern "C" void kernel_launch(void* const* d_in, const int* in_sizes, int n_in,
                              void* d_out, int out_size, void* d_ws, size_t ws_size,
                              hipStream_t stream) {
  (void)in_sizes; (void)n_in; (void)out_size; (void)ws_size;
  const float* x   = (const float*)d_in[0];
  const float* wmu = (const float*)d_in[1];
  const float* wls = (const float*)d_in[2];
  const float* bmu = (const float*)d_in[3];
  const float* bls = (const float*)d_in[4];
  const float* ew  = (const float*)d_in[5];
  const float* eb  = (const float*)d_in[6];

  float* out = (float*)d_out;
  float* unc = out + (size_t)B_SZ * OUT_SZ;

  char* ws = (char*)d_ws;
  __bf16* Xb  = (__bf16*)ws;  ws += (size_t)B_SZ * IN_SZ * sizeof(__bf16);
  __bf16* X2b = (__bf16*)ws;  ws += (size_t)B_SZ * IN_SZ * sizeof(__bf16);
  __bf16* Wb  = (__bf16*)ws;  ws += (size_t)OUT_SZ * IN_SZ * sizeof(__bf16);
  __bf16* Sb  = (__bf16*)ws;

  bayes_prep_x<<<(B_SZ * IN_SZ) / (8 * 256), 256, 0, stream>>>(x, Xb, X2b);
  bayes_prep_w<<<(OUT_SZ * IN_SZ) / (8 * 256), 256, 0, stream>>>(wmu, wls, ew, Wb, Sb);

  dim3 grid(OUT_SZ / BN, B_SZ / BM);   // 32 x 32 = 1024 workgroups
  bayes_gemm<<<grid, 256, 0, stream>>>(Xb, X2b, Wb, Sb, bmu, bls, eb, out, unc);
}
